// GATLayer_28080496181833
// MI455X (gfx1250) — matrix operand order repaired and checked by reading, not compiled, not run on hardware
//
#include <hip/hip_runtime.h>
#include <hip/hip_bf16.h>

// ---------------------------------------------------------------------------
// GAT layer for MI455X (gfx1250, wave32).
// Key algebraic optimization: e_proj is only consumed by the att_edge dot, so
// a_edge = ea @ w_eh with w_eh[d][h] = sum_c W_edge[d, h*64+c]*att_edge[h,c].
// Dense work (x_proj + residual) fused into one f16-WMMA GEMM [N,128]x[128,320].
// ---------------------------------------------------------------------------

static constexpr int IN_F  = 128;
static constexpr int OUTC  = 64;
static constexpr int NHEAD = 4;
static constexpr int EDIM  = 64;
static constexpr int WCOLS = 320;   // 256 (H*OUT proj) + 64 (residual)
static constexpr float SLOPE = 0.2f;
static constexpr float EPSF  = 1e-16f;

typedef __attribute__((ext_vector_type(16))) _Float16 v16h;
typedef __attribute__((ext_vector_type(8)))  _Float16 v8h;
typedef __attribute__((ext_vector_type(8)))  float    v8f;

__device__ __forceinline__ float lrelu(float x) { return x >= 0.f ? x : SLOPE * x; }
// Monotonic float->uint encoding for atomicMax-based segment max.
__device__ __forceinline__ unsigned encf(float x) {
    unsigned u = __float_as_uint(x);
    return (u & 0x80000000u) ? ~u : (u | 0x80000000u);
}
__device__ __forceinline__ float decf(unsigned e) {
    unsigned u = (e & 0x80000000u) ? (e ^ 0x80000000u) : ~e;
    return __uint_as_float(u);
}
static constexpr unsigned ENC_NEG_INF = 0x007FFFFFu;  // encf(-inf)

// --------------------------- weight preparation ----------------------------
__global__ void k_pack_weights(const float* __restrict__ Wlin, const float* __restrict__ Wres,
                               const float* __restrict__ Wedge, const float* __restrict__ attE,
                               _Float16* __restrict__ WhT, float* __restrict__ w_eh) {
    const int t = threadIdx.x;
    // WhT[col][k] (transposed, f16) : col<256 -> W_lin, else W_res
    for (int idx = t; idx < WCOLS * IN_F; idx += blockDim.x) {
        int col = idx / IN_F, k = idx % IN_F;
        float v = (col < 256) ? Wlin[k * 256 + col] : Wres[k * OUTC + (col - 256)];
        WhT[idx] = (_Float16)v;
    }
    // w_eh[d][h] = sum_c W_edge[d, h*64+c] * att_edge[h, c]
    if (t < EDIM * NHEAD) {
        int d = t / NHEAD, h = t % NHEAD;
        float s = 0.f;
        for (int c = 0; c < OUTC; ++c)
            s += Wedge[d * (NHEAD * OUTC) + h * OUTC + c] * attE[h * OUTC + c];
        w_eh[t] = s;
    }
}

__global__ void k_convert_x(const float* __restrict__ X, _Float16* __restrict__ Xh, int total) {
    for (int i = blockIdx.x * blockDim.x + threadIdx.x; i < total; i += gridDim.x * blockDim.x)
        Xh[i] = (_Float16)X[i];
}

__global__ void k_init(float* ea_sum, float* cnt, float* denom, unsigned* amax, int n) {
    int i = blockIdx.x * blockDim.x + threadIdx.x;
    if (i < n * EDIM) ea_sum[i] = 0.f;
    if (i < n)        cnt[i] = 0.f;
    if (i < n * NHEAD) { denom[i] = 0.f; amax[i] = ENC_NEG_INF; }
}

// ------------------------------- WMMA GEMM ---------------------------------
// C[N,320] = Xh[N,128] (f16) x W[128,320] (f16, stored transposed WhT[320][128]).
// Block: 256 thr = 8 waves; block tile 128 rows x 64 cols; K = 128 (4 x 32).
// Wave w owns row sub-tile w (16 rows) and all 4 col sub-tiles -> one A-frag
// load feeds 4 WMMAs; A tile re-read from HBM only 5x (320/64) instead of 20x.
__global__ __launch_bounds__(256) void k_gemm_wmma(const _Float16* __restrict__ Xh,
                                                   const _Float16* __restrict__ WhT,
                                                   float* __restrict__ xproj, int n) {
    __shared__ __align__(16) _Float16 Ah[128][136];   // padded: 272B row stride
    __shared__ __align__(32) _Float16 Bt[64][128];    // Bt[col][k]
    const int t = threadIdx.x;
    const int rowBase = blockIdx.x * 128;
    const int colBase = blockIdx.y * 64;
    const bool fullBlk = (rowBase + 128 <= n);

    // Stage A tile (128x128 f16) -> LDS, 16B vector copies.
    if (fullBlk) {
#pragma unroll
        for (int it = 0; it < 8; ++it) {
            int ci = t + it * 256;            // 2048 chunks of 8 halfs
            int r = ci >> 4, cc = ci & 15;
            *(uint4*)&Ah[r][cc * 8] =
                *(const uint4*)(Xh + (size_t)(rowBase + r) * IN_F + cc * 8);
        }
    } else {
#pragma unroll
        for (int it = 0; it < 8; ++it) {
            int ci = t + it * 256;
            int r = ci >> 4, cc = ci & 15;
            int gr = rowBase + r;
            uint4 v = make_uint4(0u, 0u, 0u, 0u);
            if (gr < n) v = *(const uint4*)(Xh + (size_t)gr * IN_F + cc * 8);
            *(uint4*)&Ah[r][cc * 8] = v;
        }
    }
    // Stage B tile (64 cols x 128 k) from pre-transposed weights.
#pragma unroll
    for (int it = 0; it < 4; ++it) {
        int ci = t + it * 256;                // 1024 chunks of 8 halfs
        int nl = ci >> 4, cc = ci & 15;
        *(uint4*)&Bt[nl][cc * 8] =
            *(const uint4*)(WhT + (size_t)(colBase + nl) * IN_F + cc * 8);
    }
    __syncthreads();

    const int w  = t >> 5;                // wave id: row sub-tile
    const int ln = t & 31;
    const int hl = ln >> 4, lr = ln & 15;

    v8f acc0 = {}, acc1 = {}, acc2 = {}, acc3 = {};
#pragma unroll
    for (int kk = 0; kk < 4; ++kk) {
        const int kb = kk * 32;
        // A frag (16x32 f16): lane group hl -> K hl*16..hl*16+15
        union { v16h v; struct { v8h lo, hi; } s; } a;
        a.s.lo = *(const v8h*)&Ah[w * 16 + lr][kb + hl * 16];
        a.s.hi = *(const v8h*)&Ah[w * 16 + lr][kb + hl * 16 + 8];
        // B frags (32x16 f16): lane group hl holds K = hl*16..hl*16+15 at col lr
        v16h b0 = *(const v16h*)&Bt[ 0 + lr][kb + hl * 16];
        v16h b1 = *(const v16h*)&Bt[16 + lr][kb + hl * 16];
        v16h b2 = *(const v16h*)&Bt[32 + lr][kb + hl * 16];
        v16h b3 = *(const v16h*)&Bt[48 + lr][kb + hl * 16];
        acc0 = __builtin_amdgcn_wmma_f32_16x16x32_f16(false, a.v, false, b0, (short)0, acc0, false, false);
        acc1 = __builtin_amdgcn_wmma_f32_16x16x32_f16(false, a.v, false, b1, (short)0, acc1, false, false);
        acc2 = __builtin_amdgcn_wmma_f32_16x16x32_f16(false, a.v, false, b2, (short)0, acc2, false, false);
        acc3 = __builtin_amdgcn_wmma_f32_16x16x32_f16(false, a.v, false, b3, (short)0, acc3, false, false);
    }
    // D layout: VGPR r -> M = hl*8 + r, N = lr
    float* outRow = xproj + (size_t)(rowBase + w * 16 + hl * 8) * WCOLS + colBase + lr;
    if (fullBlk) {
#pragma unroll
        for (int r = 0; r < 8; ++r) {
            float* o = outRow + (size_t)r * WCOLS;
            o[0] = acc0[r]; o[16] = acc1[r]; o[32] = acc2[r]; o[48] = acc3[r];
        }
    } else {
#pragma unroll
        for (int r = 0; r < 8; ++r) {
            int gr = rowBase + w * 16 + hl * 8 + r;
            if (gr < n) {
                float* o = outRow + (size_t)r * WCOLS;
                o[0] = acc0[r]; o[16] = acc1[r]; o[32] = acc2[r]; o[48] = acc3[r];
            }
        }
    }
}

// --------------------------- edge pass 1 -----------------------------------
// Wave per edge: ea_sum/cnt scatter (for self-loop mean attr) + a_edge = ea @ w_eh.
__global__ __launch_bounds__(256) void k_edge1(const int* __restrict__ ei,
                                               const float* __restrict__ ef,
                                               const float* __restrict__ impPtr,
                                               const float* __restrict__ w_eh,
                                               float* ea_sum, float* cnt,
                                               float* __restrict__ aeRaw, int E) {
    const int wv = threadIdx.x >> 5, ln = threadIdx.x & 31;
    const int e = blockIdx.x * 8 + wv;
    if (e >= E) return;
    const float imp = impPtr[0];
    const int d = ei[E + e];
    const int c0 = ln * 2;
    float2 v = *(const float2*)(ef + (size_t)e * EDIM + c0);
    v.x *= imp; v.y *= imp;
    atomicAdd(&ea_sum[(size_t)d * EDIM + c0], v.x);
    atomicAdd(&ea_sum[(size_t)d * EDIM + c0 + 1], v.y);
    if (ln == 0) atomicAdd(&cnt[d], 1.0f);
    const float4 w0 = *(const float4*)(w_eh + c0 * NHEAD);
    const float4 w1 = *(const float4*)(w_eh + (c0 + 1) * NHEAD);
    float p0 = v.x * w0.x + v.y * w1.x;
    float p1 = v.x * w0.y + v.y * w1.y;
    float p2 = v.x * w0.z + v.y * w1.z;
    float p3 = v.x * w0.w + v.y * w1.w;
#pragma unroll
    for (int off = 16; off >= 1; off >>= 1) {
        p0 += __shfl_xor(p0, off, 32);
        p1 += __shfl_xor(p1, off, 32);
        p2 += __shfl_xor(p2, off, 32);
        p3 += __shfl_xor(p3, off, 32);
    }
    if (ln == 0) *(float4*)(aeRaw + (size_t)e * NHEAD) = make_float4(p0, p1, p2, p3);
}

// --------------------------- node pass -------------------------------------
// Wave per node: a_src/a_dst dots and the self-loop attention logit.
__global__ __launch_bounds__(256) void k_node(const float* __restrict__ xproj,
                                              const float* __restrict__ attS,
                                              const float* __restrict__ attD,
                                              const float* __restrict__ ea_sum,
                                              const float* __restrict__ cnt,
                                              const float* __restrict__ w_eh,
                                              float* __restrict__ aSrc, float* __restrict__ aDst,
                                              float* __restrict__ logits, int n, int E) {
    const int wv = threadIdx.x >> 5, ln = threadIdx.x & 31;
    const int nd = blockIdx.x * 8 + wv;
    if (nd >= n) return;
    const int c0 = ln * 2;
    float ps[NHEAD], pd[NHEAD], pe[NHEAD];
#pragma unroll
    for (int h = 0; h < NHEAD; ++h) {
        float2 x  = *(const float2*)(xproj + (size_t)nd * WCOLS + h * OUTC + c0);
        float2 as = *(const float2*)(attS + h * OUTC + c0);
        float2 ad = *(const float2*)(attD + h * OUTC + c0);
        ps[h] = x.x * as.x + x.y * as.y;
        pd[h] = x.x * ad.x + x.y * ad.y;
    }
    float cn = cnt[nd]; cn = cn > 1.f ? cn : 1.f;
    float2 s2 = *(const float2*)(ea_sum + (size_t)nd * EDIM + c0);
    s2.x /= cn; s2.y /= cn;
    const float4 w0 = *(const float4*)(w_eh + c0 * NHEAD);
    const float4 w1 = *(const float4*)(w_eh + (c0 + 1) * NHEAD);
    pe[0] = s2.x * w0.x + s2.y * w1.x;
    pe[1] = s2.x * w0.y + s2.y * w1.y;
    pe[2] = s2.x * w0.z + s2.y * w1.z;
    pe[3] = s2.x * w0.w + s2.y * w1.w;
#pragma unroll
    for (int off = 16; off >= 1; off >>= 1) {
#pragma unroll
        for (int h = 0; h < NHEAD; ++h) {
            ps[h] += __shfl_xor(ps[h], off, 32);
            pd[h] += __shfl_xor(pd[h], off, 32);
            pe[h] += __shfl_xor(pe[h], off, 32);
        }
    }
    if (ln == 0) {
        *(float4*)(aSrc + (size_t)nd * NHEAD) = make_float4(ps[0], ps[1], ps[2], ps[3]);
        *(float4*)(aDst + (size_t)nd * NHEAD) = make_float4(pd[0], pd[1], pd[2], pd[3]);
        float4 lg;
        lg.x = lrelu(ps[0] + pd[0] + pe[0]);
        lg.y = lrelu(ps[1] + pd[1] + pe[1]);
        lg.z = lrelu(ps[2] + pd[2] + pe[2]);
        lg.w = lrelu(ps[3] + pd[3] + pe[3]);
        *(float4*)(logits + (size_t)(E + nd) * NHEAD) = lg;
    }
}

// --------------------------- segment softmax -------------------------------
__global__ void k_emax(const int* __restrict__ ei, const float* __restrict__ aSrc,
                       const float* __restrict__ aDst, const float* __restrict__ aeRaw,
                       float* __restrict__ logits, unsigned* amax, int E, int n) {
    const int i = blockIdx.x * blockDim.x + threadIdx.x;
    if (i >= E + n) return;
    int d;
    float4 lg;
    if (i < E) {
        int s = ei[i]; d = ei[E + i];
        float4 as = *(const float4*)(aSrc + (size_t)s * NHEAD);
        float4 ad = *(const float4*)(aDst + (size_t)d * NHEAD);
        float4 ae = *(const float4*)(aeRaw + (size_t)i * NHEAD);
        lg.x = lrelu(as.x + ad.x + ae.x);
        lg.y = lrelu(as.y + ad.y + ae.y);
        lg.z = lrelu(as.z + ad.z + ae.z);
        lg.w = lrelu(as.w + ad.w + ae.w);
        *(float4*)(logits + (size_t)i * NHEAD) = lg;
    } else {
        d = i - E;
        lg = *(const float4*)(logits + (size_t)i * NHEAD);
    }
    atomicMax(&amax[(size_t)d * NHEAD + 0], encf(lg.x));
    atomicMax(&amax[(size_t)d * NHEAD + 1], encf(lg.y));
    atomicMax(&amax[(size_t)d * NHEAD + 2], encf(lg.z));
    atomicMax(&amax[(size_t)d * NHEAD + 3], encf(lg.w));
}

__global__ void k_eexp(const int* __restrict__ ei, const unsigned* __restrict__ amax,
                       float* __restrict__ logits, float* denom, int E, int n) {
    const int i = blockIdx.x * blockDim.x + threadIdx.x;
    if (i >= E + n) return;
    const int d = (i < E) ? ei[E + i] : (i - E);
    float4 lg = *(const float4*)(logits + (size_t)i * NHEAD);
    float4 w;
    w.x = __expf(lg.x - decf(amax[(size_t)d * NHEAD + 0]));
    w.y = __expf(lg.y - decf(amax[(size_t)d * NHEAD + 1]));
    w.z = __expf(lg.z - decf(amax[(size_t)d * NHEAD + 2]));
    w.w = __expf(lg.w - decf(amax[(size_t)d * NHEAD + 3]));
    *(float4*)(logits + (size_t)i * NHEAD) = w;
    atomicAdd(&denom[(size_t)d * NHEAD + 0], w.x);
    atomicAdd(&denom[(size_t)d * NHEAD + 1], w.y);
    atomicAdd(&denom[(size_t)d * NHEAD + 2], w.z);
    atomicAdd(&denom[(size_t)d * NHEAD + 3], w.w);
}

// out = residual + bias (accumulation target for the edge scatter).
__global__ void k_initout(const float* __restrict__ xproj, const float* __restrict__ bias,
                          float* __restrict__ out, int n) {
    const int i = blockIdx.x * blockDim.x + threadIdx.x;
    if (i >= n * OUTC) return;
    const int nd = i / OUTC, c = i % OUTC;
    out[i] = xproj[(size_t)nd * WCOLS + 256 + c] + bias[c];
}

// --------------------------- message aggregation ---------------------------
// Wave per edge: gather 256-float x_proj row, fold head-mean (x0.25) into the
// per-head softmax coefficient, scatter 64 atomic f32 adds into out.
__global__ __launch_bounds__(256) void k_eagg(const int* __restrict__ ei,
                                              const float* __restrict__ logits,
                                              const float* __restrict__ denom,
                                              const float* __restrict__ xproj,
                                              float* out, int E, int n) {
    const int wv = threadIdx.x >> 5, ln = threadIdx.x & 31;
    const int i = blockIdx.x * 8 + wv;
    if (i >= E + n) return;
    int s, d;
    if (i < E) { s = ei[i]; d = ei[E + i]; } else { s = d = i - E; }
    float4 lg = *(const float4*)(logits + (size_t)i * NHEAD);
    float4 dn = *(const float4*)(denom + (size_t)d * NHEAD);
    const float c0 = lg.x / (dn.x + EPSF) * 0.25f;
    const float c1 = lg.y / (dn.y + EPSF) * 0.25f;
    const float c2 = lg.z / (dn.z + EPSF) * 0.25f;
    const float c3 = lg.w / (dn.w + EPSF) * 0.25f;
    const int cc = ln * 2;
    const float* xr = xproj + (size_t)s * WCOLS;
    float2 x0 = *(const float2*)(xr + 0 * OUTC + cc);
    float2 x1 = *(const float2*)(xr + 1 * OUTC + cc);
    float2 x2 = *(const float2*)(xr + 2 * OUTC + cc);
    float2 x3 = *(const float2*)(xr + 3 * OUTC + cc);
    float rx = c0 * x0.x + c1 * x1.x + c2 * x2.x + c3 * x3.x;
    float ry = c0 * x0.y + c1 * x1.y + c2 * x2.y + c3 * x3.y;
    atomicAdd(&out[(size_t)d * OUTC + cc], rx);
    atomicAdd(&out[(size_t)d * OUTC + cc + 1], ry);
}

// ---------------------------------------------------------------------------
extern "C" void kernel_launch(void* const* d_in, const int* in_sizes, int n_in,
                              void* d_out, int out_size, void* d_ws, size_t ws_size,
                              hipStream_t stream) {
    const float* X     = (const float*)d_in[0];
    const int*   ei    = (const int*)d_in[1];
    const float* ef    = (const float*)d_in[2];
    const float* imp   = (const float*)d_in[3];
    const float* Wlin  = (const float*)d_in[4];
    const float* attS  = (const float*)d_in[5];
    const float* attD  = (const float*)d_in[6];
    const float* Wedge = (const float*)d_in[7];
    const float* attE  = (const float*)d_in[8];
    const float* Wres  = (const float*)d_in[9];
    const float* bias  = (const float*)d_in[10];
    float* out = (float*)d_out;

    const int n  = in_sizes[0] / IN_F;
    const int E  = in_sizes[1] / 2;
    const int EN = E + n;

    char* p = (char*)d_ws;
    auto alloc = [&](size_t bytes) -> char* {
        char* r = p; p += (bytes + 255) & ~(size_t)255; return r;
    };
    _Float16* Xh   = (_Float16*)alloc((size_t)n * IN_F * 2);
    _Float16* WhT  = (_Float16*)alloc((size_t)WCOLS * IN_F * 2);
    float* w_eh    = (float*)alloc((size_t)EDIM * NHEAD * 4);
    float* xproj   = (float*)alloc((size_t)n * WCOLS * 4);
    float* ea_sum  = (float*)alloc((size_t)n * EDIM * 4);
    float* cnt     = (float*)alloc((size_t)n * 4);
    float* aSrc    = (float*)alloc((size_t)n * NHEAD * 4);
    float* aDst    = (float*)alloc((size_t)n * NHEAD * 4);
    float* aeRaw   = (float*)alloc((size_t)E * NHEAD * 4);
    float* logits  = (float*)alloc((size_t)EN * NHEAD * 4);
    unsigned* amax = (unsigned*)alloc((size_t)n * NHEAD * 4);
    float* denom   = (float*)alloc((size_t)n * NHEAD * 4);

    k_pack_weights<<<1, 256, 0, stream>>>(Wlin, Wres, Wedge, attE, WhT, w_eh);
    k_convert_x<<<2048, 256, 0, stream>>>(X, Xh, n * IN_F);
    k_init<<<(n * EDIM + 255) / 256, 256, 0, stream>>>(ea_sum, cnt, denom, amax, n);

    dim3 gg((n + 127) / 128, WCOLS / 64);
    k_gemm_wmma<<<gg, 256, 0, stream>>>(Xh, WhT, xproj, n);

    k_edge1<<<(E + 7) / 8, 256, 0, stream>>>(ei, ef, imp, w_eh, ea_sum, cnt, aeRaw, E);
    k_node<<<(n + 7) / 8, 256, 0, stream>>>(xproj, attS, attD, ea_sum, cnt, w_eh,
                                            aSrc, aDst, logits, n, E);
    k_emax<<<(EN + 255) / 256, 256, 0, stream>>>(ei, aSrc, aDst, aeRaw, logits, amax, E, n);
    k_eexp<<<(EN + 255) / 256, 256, 0, stream>>>(ei, amax, logits, denom, E, n);
    k_initout<<<(n * OUTC + 255) / 256, 256, 0, stream>>>(xproj, bias, out, n);
    k_eagg<<<(EN + 7) / 8, 256, 0, stream>>>(ei, logits, denom, xproj, out, E, n);
}
